// MaskedMoE2_30897994727627
// MI455X (gfx1250) — matrix operand order repaired and checked by reading, not compiled, not run on hardware
//
#include <hip/hip_runtime.h>
#include <hip/hip_bf16.h>
#include <math.h>
#include <stdint.h>

// Problem constants (match reference)
#define TT   4096      // B*S tokens
#define DD   1024      // model dim
#define DFFC 4096      // ffn dim
#define EE   8         // experts
#define EP1  9         // experts + dummy
#define KSEL 2         // top_k

// GEMM tile
#define BM 64
#define BN 128
#define BK 32

typedef __attribute__((ext_vector_type(16))) __bf16 v16bf;
typedef __attribute__((ext_vector_type(8)))  __bf16 v8bf;
typedef __attribute__((ext_vector_type(8)))  float  v8f;
typedef __attribute__((ext_vector_type(4)))  unsigned int u32x4;
typedef __attribute__((ext_vector_type(8)))  unsigned int u32x8;

__device__ __forceinline__ float gelu_exact(float x) {
    return 0.5f * x * (1.0f + erff(x * 0.7071067811865475f));
}

// Low 32 bits of a generic pointer to LDS == wave-relative LDS byte address (ISA 10.2)
__device__ __forceinline__ unsigned ldsaddr(const void* p) {
    return (unsigned)(unsigned long long)(uintptr_t)p;
}

// ---- CDNA5 async data movers (inline asm; CDNA5 ISA) ----

// Per-lane 16B global -> LDS, tracked by ASYNCcnt
__device__ __forceinline__ void async_load_b128(unsigned lds_off, const void* gptr) {
    asm volatile("global_load_async_to_lds_b128 %0, %1, off"
                 :: "v"(lds_off), "v"((unsigned long long)(uintptr_t)gptr)
                 : "memory");
}

__device__ __forceinline__ void wait_asynccnt0() {
    asm volatile("s_wait_asynccnt 0x0" ::: "memory");
}

__device__ __forceinline__ void wait_tensorcnt0() {
#if __has_builtin(__builtin_amdgcn_s_wait_tensorcnt)
    __builtin_amdgcn_s_wait_tensorcnt(0);
#else
    asm volatile("s_wait_tensorcnt 0x0" ::: "memory");
#endif
}

// TDM: 2D tile (tile_rows x row_u64*8 bytes) global -> LDS, one DMA per issue.
// D# layout per CDNA5 ISA; data_size = 8B units.
__device__ __forceinline__ void tdm_load_2d(unsigned lds_off, const void* gptr,
                                            unsigned tile_rows, unsigned row_u64,
                                            unsigned stride_u64, unsigned tensor_d0,
                                            unsigned tensor_d1) {
    unsigned long long ga = (unsigned long long)(uintptr_t)gptr;
    u32x4 g0;
    g0[0] = 1u;                                              // count=1, user descriptor
    g0[1] = lds_off;                                         // lds_addr (bytes)
    g0[2] = (unsigned)ga;                                    // global_addr[31:0]
    g0[3] = (unsigned)((ga >> 32) & 0x01FFFFFFu) | 0x80000000u; // addr[56:32] | type=2
    u32x8 g1;
    g1[0] = (3u << 16);                                      // data_size = 8 bytes
    g1[1] = (tensor_d0 & 0xFFFFu) << 16;                     // tensor_dim0[15:0]
    g1[2] = (tensor_d0 >> 16) | ((tensor_d1 & 0xFFFFu) << 16);
    g1[3] = (tensor_d1 >> 16) | (row_u64 << 16);             // tile_dim0
    g1[4] = tile_rows & 0xFFFFu;                             // tile_dim1 (tile_dim2=0)
    g1[5] = stride_u64;                                      // tensor_dim0_stride[31:0]
    g1[6] = 0u;                                              // stride hi / dim1_stride lo
    g1[7] = 0u;
    asm volatile("tensor_load_to_lds %0, %1" :: "s"(g0), "s"(g1) : "memory");
}

// ---------------------------------------------------------------- utilities
__global__ void k_zero_counts(int* counts) {
    if (threadIdx.x < EE) counts[threadIdx.x] = 0;
}

// fp32 -> bf16, 8 elements per thread
__global__ void k_cvt_x(const float* __restrict__ in, __bf16* __restrict__ out, int n8) {
    int i = blockIdx.x * blockDim.x + threadIdx.x;
    if (i >= n8) return;
    const float4* p = (const float4*)in + (size_t)i * 2;
    float4 a = p[0], b = p[1];
    v8bf o;
    o[0] = (__bf16)a.x; o[1] = (__bf16)a.y; o[2] = (__bf16)a.z; o[3] = (__bf16)a.w;
    o[4] = (__bf16)b.x; o[5] = (__bf16)b.y; o[6] = (__bf16)b.z; o[7] = (__bf16)b.w;
    *((v8bf*)out + i) = o;
}

// [E][R][C] fp32 -> [E][C][R] bf16 (transpose + convert), 32x32 tiles
__global__ __launch_bounds__(256)
void k_tcvt(const float* __restrict__ in, __bf16* __restrict__ out, int R, int C) {
    __shared__ float tile[32][33];
    const int e  = blockIdx.z;
    const int c0 = blockIdx.x * 32;
    const int r0 = blockIdx.y * 32;
    const int tx = threadIdx.x;   // 0..31
    const int ty = threadIdx.y;   // 0..7
    const float* src = in + (size_t)e * R * C;
    __bf16*      dst = out + (size_t)e * C * R;
#pragma unroll
    for (int i = 0; i < 4; ++i)
        tile[ty + i * 8][tx] = src[(size_t)(r0 + ty + i * 8) * C + (c0 + tx)];
    __syncthreads();
#pragma unroll
    for (int i = 0; i < 4; ++i)
        dst[(size_t)(c0 + ty + i * 8) * R + (r0 + tx)] = (__bf16)tile[tx][ty + i * 8];
}

// ---------------------------------------------------------------- router
__global__ void k_router(const float* __restrict__ xs,
                         const float* __restrict__ masks,
                         const float* __restrict__ rw,
                         float* __restrict__ dout,
                         int* counts, int* btok, int* bslot, float* bgate) {
    const int t    = (blockIdx.x * blockDim.x + threadIdx.x) >> 5;
    const int lane = threadIdx.x & 31;
    if (t >= TT) return;
    const float* x = xs + (size_t)t * DD;
    float acc[EP1] = {};
    for (int i = lane; i < DD; i += 32) {
        float xv = x[i];
#pragma unroll
        for (int e = 0; e < EP1; ++e) acc[e] = fmaf(xv, rw[e * DD + i], acc[e]);
    }
#pragma unroll
    for (int e = 0; e < EP1; ++e) {
#pragma unroll
        for (int off = 16; off > 0; off >>= 1)
            acc[e] += __shfl_xor(acc[e], off, 32);
    }
    if (lane != 0) return;

    float lg[EP1];
#pragma unroll
    for (int e = 0; e < EE; ++e) lg[e] = acc[e] * masks[(size_t)t * EE + e];
    lg[EE] = acc[EE];

    float* lout = dout + (size_t)TT * DD + (size_t)t * EP1;
#pragma unroll
    for (int e = 0; e < EP1; ++e) lout[e] = lg[e];

    float mx = lg[0];
#pragma unroll
    for (int e = 1; e < EP1; ++e) mx = fmaxf(mx, lg[e]);
    float p[EP1], s = 0.f;
#pragma unroll
    for (int e = 0; e < EP1; ++e) { p[e] = __expf(lg[e] - mx); s += p[e]; }
    float inv = 1.f / s;
#pragma unroll
    for (int e = 0; e < EP1; ++e) p[e] *= inv;

    int i1 = 0; float v1 = p[0];
#pragma unroll
    for (int e = 1; e < EP1; ++e) if (p[e] > v1) { v1 = p[e]; i1 = e; }
    int i2 = -1; float v2 = -1.f;
#pragma unroll
    for (int e = 0; e < EP1; ++e) if (e != i1 && p[e] > v2) { v2 = p[e]; i2 = e; }

    int* selo = (int*)dout + (size_t)TT * DD + (size_t)TT * EP1 + (size_t)t * KSEL;
    selo[0] = i1; selo[1] = i2;

    int   sel[2] = { i1, i2 };
    float wv[2]  = { v1, v2 };
#pragma unroll
    for (int j = 0; j < KSEL; ++j) {
        int e = sel[j];
        if (e < EE) {
            int pos = atomicAdd(&counts[e], 1);
            btok[e * TT + pos]  = t;
            bslot[e * TT + pos] = t * KSEL + j;
            bgate[e * TT + pos] = wv[j];
        }
    }
}

// ---------------------------------------------------------------- FFN GEMM 1
// h[slot] = gelu(x[tok] @ w1[e] + b1[e]) ; gathered rows, TDM weights, async acts.
__global__ __launch_bounds__(256)
void k_ffn1(const __bf16* __restrict__ xsb,          // [T][D] bf16
            const __bf16* __restrict__ w1t,          // [E][DFF][D] bf16 (B^T)
            const float*  __restrict__ b1,           // [E][DFF]
            const int* __restrict__ counts,
            const int* __restrict__ btok,
            const int* __restrict__ bslot,
            __bf16* __restrict__ hslot) {            // [T*K][DFF] bf16
    __shared__ __align__(32) __bf16 sA[2][BM * BK];
    __shared__ __align__(32) __bf16 sB[2][BN * BK];
    __shared__ int sTok[BM];
    __shared__ int sSlot[BM];

    const int e   = blockIdx.z;
    const int cnt = counts[e];
    const int m0  = blockIdx.y * BM;
    if (m0 >= cnt) return;
    const int n0   = blockIdx.x * BN;
    const int tid  = threadIdx.x;
    const int lane = tid & 31;
    const int wv   = tid >> 5;
    const int wm   = wv & 3;     // wave row block (16 rows)
    const int wn   = wv >> 2;    // wave col block (64 cols)

    if (tid < BM) {
        int pos = m0 + tid;
        sTok[tid]  = (pos < cnt) ? btok[e * TT + pos]  : btok[e * TT + m0];
        sSlot[tid] = (pos < cnt) ? bslot[e * TT + pos] : -1;
    }
    __syncthreads();

    const int arow = tid >> 2, ach = tid & 3;   // A: 64 rows x 4 chunks of 16B
    const size_t aoff = (size_t)sTok[arow] * DD + ach * 8;
    const __bf16* wbase = w1t + (size_t)e * DFFC * DD + (size_t)n0 * DD;
    const unsigned sAoff[2] = { ldsaddr(&sA[0][0]), ldsaddr(&sA[1][0]) };
    const unsigned sBoff[2] = { ldsaddr(&sB[0][0]), ldsaddr(&sB[1][0]) };
    const unsigned aLds = (unsigned)((arow * BK + ach * 8) * 2);

    // stage tile k0 into buffer b: A via per-lane async, B via one TDM per block
    #define STAGE1(b, k0)                                                          \
        do {                                                                       \
            async_load_b128(sAoff[b] + aLds, xsb + aoff + (k0));                   \
            if (wv == 0)                                                           \
                tdm_load_2d(sBoff[b], wbase + (k0), BN, 8u, DD / 4u, DD / 4u,      \
                            1u << 20);                                             \
        } while (0)

    v8f acc0 = {}, acc1 = {}, acc2 = {}, acc3 = {};
    STAGE1(0, 0);
    wait_asynccnt0();
    wait_tensorcnt0();
    __syncthreads();

    int buf = 0;
    for (int k0 = 0; k0 < DD; k0 += BK) {
        if (k0 + BK < DD) STAGE1(buf ^ 1, k0 + BK);

        const int r  = lane & 15;
        const int sh = (lane >> 4) * 16;
        const __bf16* pa = &sA[buf][(wm * 16 + r) * BK];
        v8bf alo = *(const v8bf*)(pa + sh);
        v8bf ahi = *(const v8bf*)(pa + sh + 8);
        v16bf af = __builtin_shufflevector(alo, ahi, 0,1,2,3,4,5,6,7,8,9,10,11,12,13,14,15);

        const int kb = (lane >> 4) * 16;
        const int nb = wn * 64 + (lane & 15);
        v16bf bf0 = *(const v16bf*)&sB[buf][(nb +  0) * BK + kb];
        v16bf bf1 = *(const v16bf*)&sB[buf][(nb + 16) * BK + kb];
        v16bf bf2 = *(const v16bf*)&sB[buf][(nb + 32) * BK + kb];
        v16bf bf3 = *(const v16bf*)&sB[buf][(nb + 48) * BK + kb];
        acc0 = __builtin_amdgcn_wmma_f32_16x16x32_bf16(false, af, false, bf0, (short)0, acc0, false, false);
        acc1 = __builtin_amdgcn_wmma_f32_16x16x32_bf16(false, af, false, bf1, (short)0, acc1, false, false);
        acc2 = __builtin_amdgcn_wmma_f32_16x16x32_bf16(false, af, false, bf2, (short)0, acc2, false, false);
        acc3 = __builtin_amdgcn_wmma_f32_16x16x32_bf16(false, af, false, bf3, (short)0, acc3, false, false);

        wait_asynccnt0();
        wait_tensorcnt0();
        __syncthreads();
        buf ^= 1;
    }
    #undef STAGE1

    const int mhi  = (lane >> 4) * 8;
    const int ncol = lane & 15;
    const float* b1e = b1 + (size_t)e * DFFC;
#pragma unroll
    for (int v = 0; v < 8; ++v) {
        int ml = wm * 16 + v + mhi;
        int slot = sSlot[ml];
        if (slot < 0) continue;
        size_t base = (size_t)slot * DFFC;
        int gn = n0 + wn * 64 + ncol;
        hslot[base + gn +  0] = (__bf16)gelu_exact(acc0[v] + b1e[gn +  0]);
        hslot[base + gn + 16] = (__bf16)gelu_exact(acc1[v] + b1e[gn + 16]);
        hslot[base + gn + 32] = (__bf16)gelu_exact(acc2[v] + b1e[gn + 32]);
        hslot[base + gn + 48] = (__bf16)gelu_exact(acc3[v] + b1e[gn + 48]);
    }
}

// ---------------------------------------------------------------- FFN GEMM 2
// out[slot] = gate * (h[slot] @ w2[e] + b2[e])
__global__ __launch_bounds__(256)
void k_ffn2(const __bf16* __restrict__ hslot,        // [T*K][DFF] bf16
            const __bf16* __restrict__ w2t,          // [E][D][DFF] bf16 (B^T)
            const float*  __restrict__ b2,           // [E][D]
            const int* __restrict__ counts,
            const int* __restrict__ bslot,
            const float* __restrict__ bgate,
            float* __restrict__ oslot) {             // [T*K][D] f32
    __shared__ __align__(32) __bf16 sA[2][BM * BK];
    __shared__ __align__(32) __bf16 sB[2][BN * BK];
    __shared__ int   sSlot[BM];
    __shared__ float sGate[BM];

    const int e   = blockIdx.z;
    const int cnt = counts[e];
    const int m0  = blockIdx.y * BM;
    if (m0 >= cnt) return;
    const int n0   = blockIdx.x * BN;
    const int tid  = threadIdx.x;
    const int lane = tid & 31;
    const int wv   = tid >> 5;
    const int wm   = wv & 3;
    const int wn   = wv >> 2;

    if (tid < BM) {
        int pos = m0 + tid;
        sSlot[tid] = (pos < cnt) ? bslot[e * TT + pos] : -1;
        sGate[tid] = (pos < cnt) ? bgate[e * TT + pos] : 0.f;
    }
    __syncthreads();

    const int arow = tid >> 2, ach = tid & 3;
    int asl = sSlot[arow]; if (asl < 0) asl = 0;      // padded rows: harmless reads
    const size_t aoff = (size_t)asl * DFFC + ach * 8;
    const __bf16* wbase = w2t + (size_t)e * DD * DFFC + (size_t)n0 * DFFC;
    const unsigned sAoff[2] = { ldsaddr(&sA[0][0]), ldsaddr(&sA[1][0]) };
    const unsigned sBoff[2] = { ldsaddr(&sB[0][0]), ldsaddr(&sB[1][0]) };
    const unsigned aLds = (unsigned)((arow * BK + ach * 8) * 2);

    #define STAGE2(b, k0)                                                           \
        do {                                                                        \
            async_load_b128(sAoff[b] + aLds, hslot + aoff + (k0));                  \
            if (wv == 0)                                                            \
                tdm_load_2d(sBoff[b], wbase + (k0), BN, 8u, DFFC / 4u, DFFC / 4u,   \
                            1u << 20);                                              \
        } while (0)

    v8f acc0 = {}, acc1 = {}, acc2 = {}, acc3 = {};
    STAGE2(0, 0);
    wait_asynccnt0();
    wait_tensorcnt0();
    __syncthreads();

    int buf = 0;
    for (int k0 = 0; k0 < DFFC; k0 += BK) {
        if (k0 + BK < DFFC) STAGE2(buf ^ 1, k0 + BK);

        const int r  = lane & 15;
        const int sh = (lane >> 4) * 16;
        const __bf16* pa = &sA[buf][(wm * 16 + r) * BK];
        v8bf alo = *(const v8bf*)(pa + sh);
        v8bf ahi = *(const v8bf*)(pa + sh + 8);
        v16bf af = __builtin_shufflevector(alo, ahi, 0,1,2,3,4,5,6,7,8,9,10,11,12,13,14,15);

        const int kb = (lane >> 4) * 16;
        const int nb = wn * 64 + (lane & 15);
        v16bf bf0 = *(const v16bf*)&sB[buf][(nb +  0) * BK + kb];
        v16bf bf1 = *(const v16bf*)&sB[buf][(nb + 16) * BK + kb];
        v16bf bf2 = *(const v16bf*)&sB[buf][(nb + 32) * BK + kb];
        v16bf bf3 = *(const v16bf*)&sB[buf][(nb + 48) * BK + kb];
        acc0 = __builtin_amdgcn_wmma_f32_16x16x32_bf16(false, af, false, bf0, (short)0, acc0, false, false);
        acc1 = __builtin_amdgcn_wmma_f32_16x16x32_bf16(false, af, false, bf1, (short)0, acc1, false, false);
        acc2 = __builtin_amdgcn_wmma_f32_16x16x32_bf16(false, af, false, bf2, (short)0, acc2, false, false);
        acc3 = __builtin_amdgcn_wmma_f32_16x16x32_bf16(false, af, false, bf3, (short)0, acc3, false, false);

        wait_asynccnt0();
        wait_tensorcnt0();
        __syncthreads();
        buf ^= 1;
    }
    #undef STAGE2

    const int mhi  = (lane >> 4) * 8;
    const int ncol = lane & 15;
    const float* b2e = b2 + (size_t)e * DD;
#pragma unroll
    for (int v = 0; v < 8; ++v) {
        int ml = wm * 16 + v + mhi;
        int slot = sSlot[ml];
        if (slot < 0) continue;
        float g = sGate[ml];
        size_t base = (size_t)slot * DD;
        int gn = n0 + wn * 64 + ncol;
        oslot[base + gn +  0] = g * (acc0[v] + b2e[gn +  0]);
        oslot[base + gn + 16] = g * (acc1[v] + b2e[gn + 16]);
        oslot[base + gn + 32] = g * (acc2[v] + b2e[gn + 32]);
        oslot[base + gn + 48] = g * (acc3[v] + b2e[gn + 48]);
    }
}

// ---------------------------------------------------------------- combine
__global__ void k_combine(const float* __restrict__ oslot,
                          const int*   __restrict__ selp,
                          float* __restrict__ res) {
    int idx = blockIdx.x * blockDim.x + threadIdx.x;  // over T * D/4
    const int total = TT * (DD / 4);
    if (idx >= total) return;
    int t  = idx / (DD / 4);
    int c4 = idx % (DD / 4);
    int s0 = selp[t * KSEL + 0];
    int s1 = selp[t * KSEL + 1];
    float4 r = make_float4(0.f, 0.f, 0.f, 0.f);
    if (s0 < EE) {
        float4 v = *(const float4*)&oslot[((size_t)t * KSEL + 0) * DD + c4 * 4];
        r.x += v.x; r.y += v.y; r.z += v.z; r.w += v.w;
    }
    if (s1 < EE) {
        float4 v = *(const float4*)&oslot[((size_t)t * KSEL + 1) * DD + c4 * 4];
        r.x += v.x; r.y += v.y; r.z += v.z; r.w += v.w;
    }
    *(float4*)&res[(size_t)t * DD + c4 * 4] = r;
}

// ---------------------------------------------------------------- launch
extern "C" void kernel_launch(void* const* d_in, const int* in_sizes, int n_in,
                              void* d_out, int out_size, void* d_ws, size_t ws_size,
                              hipStream_t stream) {
    const float* xs    = (const float*)d_in[0];
    const float* masks = (const float*)d_in[1];
    const float* rw    = (const float*)d_in[2];
    const float* w1    = (const float*)d_in[3];
    const float* b1    = (const float*)d_in[4];
    const float* w2    = (const float*)d_in[5];
    const float* b2    = (const float*)d_in[6];
    float* out = (float*)d_out;

    char* ws = (char*)d_ws;
    __bf16* xsb   = (__bf16*)ws;  ws += (size_t)TT * DD * 2;
    __bf16* w1t   = (__bf16*)ws;  ws += (size_t)EE * DFFC * DD * 2;
    __bf16* w2t   = (__bf16*)ws;  ws += (size_t)EE * DD * DFFC * 2;
    __bf16* hsl   = (__bf16*)ws;  ws += (size_t)TT * KSEL * DFFC * 2;
    float*  osl   = (float*)ws;   ws += (size_t)TT * KSEL * DD * 4;
    int*    cnts  = (int*)ws;     ws += 256;
    int*    btok  = (int*)ws;     ws += (size_t)EE * TT * 4;
    int*    bslot = (int*)ws;     ws += (size_t)EE * TT * 4;
    float*  bgate = (float*)ws;   ws += (size_t)EE * TT * 4;

    k_zero_counts<<<1, 32, 0, stream>>>(cnts);
    k_cvt_x<<<(TT * DD / 8 + 255) / 256, 256, 0, stream>>>(xs, xsb, TT * DD / 8);
    k_tcvt<<<dim3(DFFC / 32, DD / 32, EE), dim3(32, 8), 0, stream>>>(w1, w1t, DD, DFFC);
    k_tcvt<<<dim3(DD / 32, DFFC / 32, EE), dim3(32, 8), 0, stream>>>(w2, w2t, DFFC, DD);
    k_router<<<TT / 8, 256, 0, stream>>>(xs, masks, rw, out, cnts, btok, bslot, bgate);
    k_ffn1<<<dim3(DFFC / BN, TT / BM, EE), 256, 0, stream>>>(xsb, w1t, b1, cnts, btok, bslot, hsl);
    k_ffn2<<<dim3(DD / BN, TT / BM, EE), 256, 0, stream>>>(hsl, w2t, b2, cnts, bslot, bgate, osl);
    const int* selp = (const int*)d_out + (size_t)TT * DD + (size_t)TT * EP1;
    k_combine<<<(TT * DD / 4 + 255) / 256, 256, 0, stream>>>(osl, selp, out);
}
